// GATModel_83700322665088
// MI455X (gfx1250) — matrix operand order repaired and checked by reading, not compiled, not run on hardware
//
#include <hip/hip_runtime.h>
#include <hip/hip_bf16.h>

// ---------------------------------------------------------------------------
// GAT model on gfx1250: all GEMMs via v_wmma_f32_16x16x32_bf16 (f32 accum).
// B matrices pre-transposed to N-major so LDS staging is all 16B copies and
// WMMA B fragments are K-contiguous b128 LDS loads. GEMM is double-buffered,
// using GLOBAL_LOAD_ASYNC_TO_LDS_B128 (ASYNCcnt) when the toolchain exposes
// it; MLP0 uses split-K for parallelism.
// ---------------------------------------------------------------------------

#define Bt 512
#define Ft 128
#define NEMB 64
#define NHEAD 8
#define GHID 32
#define C1 256          // NHEAD*GHID
#define MLP_HID 256
#define DIN (Ft * C1)   // 32768

typedef __attribute__((ext_vector_type(16))) __bf16 v16bf;
typedef __attribute__((ext_vector_type(8)))  __bf16 v8bf;
typedef __attribute__((ext_vector_type(8)))  float  v8f;
typedef __attribute__((ext_vector_type(4)))  int    v4i;

union AB16 { v16bf v; v8bf h[2]; };

// --- gfx1250 async global->LDS staging (guarded: falls back to VGPR bounce) --
#if defined(__gfx1250__) && __has_builtin(__builtin_amdgcn_global_load_async_to_lds_b128)
#define HAVE_ASYNC 1
#else
#define HAVE_ASYNC 0
#endif

#if HAVE_ASYNC
typedef __attribute__((address_space(1))) v4i gv4i_t;
typedef __attribute__((address_space(3))) v4i lv4i_t;
__device__ __forceinline__ void async_copy16(const __bf16* g, __bf16* l) {
  __builtin_amdgcn_global_load_async_to_lds_b128(
      (gv4i_t*)g, (lv4i_t*)l, 0, 0);
}
__device__ __forceinline__ void async_wait0() {
#if __has_builtin(__builtin_amdgcn_s_wait_asynccnt)
  __builtin_amdgcn_s_wait_asynccnt(0);
#else
  asm volatile("s_wait_asynccnt 0x0" ::: "memory");
#endif
}
#endif

// A fragment (16x32, M x K): lane l holds row m=l&15; element e -> K = 16*(l>>4) + e.
// `row` points at &lds[m*ld + k0].
__device__ inline v16bf load_frag_A(const __bf16* row, int lane) {
  const int kh = (lane >> 4) << 4;                 // 0 or 16
  AB16 u;
  u.h[0] = *(const v8bf*)(row + kh);
  u.h[1] = *(const v8bf*)(row + kh + 8);
  return u.v;
}

// B fragment (32x16, K x N) staged N-major in LDS: lane l holds col n=l&15;
// element e -> K = 16*(l>>4) + e.  `col` points at &lds[n*ld + k0].
__device__ inline v16bf load_frag_B(const __bf16* col, int lane) {
  const int kh = (lane >> 4) << 4;                 // 0 or 16
  return *(const v16bf*)(col + kh);
}

// ---------------------------------------------------------------------------
// Prep kernels
// ---------------------------------------------------------------------------
__global__ void gather_scale_kernel(const int* __restrict__ ids,
                                    const float* __restrict__ vals,
                                    const float* __restrict__ emb,
                                    __bf16* __restrict__ xh) {
  int i = blockIdx.x * 256 + threadIdx.x;          // over B*F*NEMB
  int row = i >> 6, col = i & 63;
  xh[i] = (__bf16)(emb[(size_t)ids[row] * NEMB + col] * vals[row]);
}

// W [H, nin, 32] (f32) -> out [256, nin] (bf16, N-major), n = h*32+o
__global__ void cat_w_t_kernel(const float* __restrict__ W, __bf16* __restrict__ out, int nin) {
  int i = blockIdx.x * 256 + threadIdx.x;          // over 256*nin
  if (i >= 256 * nin) return;
  int n = i / nin;
  int r = i - n * nin;
  int h = n >> 5, o = n & 31;
  out[i] = (__bf16)W[((size_t)h * nin + r) * GHID + o];
}

// in [K,N] f32  ->  out [N,K] bf16, 64x64 LDS-tiled transpose (coalesced both sides)
__global__ __launch_bounds__(256)
void trans_f2bf_kernel(const float* __restrict__ in, __bf16* __restrict__ out, int K, int N) {
  __shared__ float tile[64][65];
  const int kb = blockIdx.x * 64, nb = blockIdx.y * 64;
  const int t = threadIdx.x;
  const int lr = t >> 2;             // 0..63
  const int lc = (t & 3) * 16;       // 0,16,32,48
  #pragma unroll
  for (int e = 0; e < 16; ++e)
    tile[lr][lc + e] = in[(size_t)(kb + lr) * N + nb + lc + e];
  __syncthreads();
  #pragma unroll
  for (int e = 0; e < 16; ++e)
    out[(size_t)(nb + lr) * K + kb + lc + e] = (__bf16)tile[lc + e][lr];
}

// ---------------------------------------------------------------------------
// Tiled bf16 GEMM, B pre-transposed ([N,K] row-major).
// Block = 8 waves, 64(M) x 128(N) tile; wave grid 2x4, each wave a 32x32 tile
// (2 A-frags x 2 B-frags -> 4 WMMA per 32-K chunk). Double-buffered LDS with
// async global->LDS staging when available. Optional split-K via gridDim.z:
// slice s covers K [s*ksplit, (s+1)*ksplit), f32 partials at Cf + s*M*N.
// ---------------------------------------------------------------------------
#define TBM 64
#define TBN 128
#define TKC 32

template<int ACT, bool WBF, bool WF32>   // ACT: 0=none, 1=relu
__global__ __launch_bounds__(256)
void gemm_bf16_kernel(const __bf16* __restrict__ A, const __bf16* __restrict__ Bn,
                      const float* __restrict__ bias, __bf16* __restrict__ C,
                      float* __restrict__ Cf, int M, int N, int K, int ksplit) {
  __shared__ __align__(64) __bf16 sA[2][TBM * TKC];   // [m][k]
  __shared__ __align__(64) __bf16 sB[2][TBN * TKC];   // [n][k]
  const int t = threadIdx.x;
  const int lane = t & 31;
  const int wave = t >> 5;
  const int wm = wave >> 2;          // 0..1
  const int wn = wave & 3;           // 0..3
  const int m0 = blockIdx.y * TBM;
  const int n0 = blockIdx.x * TBN;
  const int kb = blockIdx.z * ksplit;
  const int ke = (kb + ksplit < K) ? kb + ksplit : K;
  const int nk = (ke - kb) / TKC;
  float* cf = WF32 ? (Cf + (size_t)blockIdx.z * M * N) : nullptr;

  auto loadA = [&](int buf, int k0) {
    int row = t >> 2;                // 0..63
    int seg = (t & 3) * 8;
    const __bf16* g = &A[(size_t)(m0 + row) * K + k0 + seg];
    __bf16* l = &sA[buf][row * TKC + seg];
#if HAVE_ASYNC
    async_copy16(g, l);
#else
    *(v8bf*)l = *(const v8bf*)g;
#endif
  };
  auto loadB = [&](int buf, int k0) {
    #pragma unroll
    for (int e = 0; e < 2; ++e) {
      int idx = t + e * 256;         // 0..511
      int n = idx >> 2;              // 0..127
      int seg = (idx & 3) * 8;
      const __bf16* g = &Bn[(size_t)(n0 + n) * K + k0 + seg];
      __bf16* l = &sB[buf][n * TKC + seg];
#if HAVE_ASYNC
      async_copy16(g, l);
#else
      *(v8bf*)l = *(const v8bf*)g;
#endif
    }
  };

  v8f c00 = {}, c01 = {}, c10 = {}, c11 = {};
  loadA(0, kb); loadB(0, kb);
  for (int i = 0; i < nk; ++i) {
#if HAVE_ASYNC
    async_wait0();                   // current chunk's async LDS writes done
#endif
    __syncthreads();
    if (i + 1 < nk) { loadA((i + 1) & 1, kb + (i + 1) * TKC); loadB((i + 1) & 1, kb + (i + 1) * TKC); }
    const __bf16* bA = sA[i & 1];
    const __bf16* bB = sB[i & 1];
    v16bf a0 = load_frag_A(&bA[(wm * 32 + 0  + (lane & 15)) * TKC], lane);
    v16bf a1 = load_frag_A(&bA[(wm * 32 + 16 + (lane & 15)) * TKC], lane);
    v16bf b0 = load_frag_B(&bB[(wn * 32 + 0  + (lane & 15)) * TKC], lane);
    v16bf b1 = load_frag_B(&bB[(wn * 32 + 16 + (lane & 15)) * TKC], lane);
    c00 = __builtin_amdgcn_wmma_f32_16x16x32_bf16(false, a0, false, b0, (short)0, c00, false, false);
    c01 = __builtin_amdgcn_wmma_f32_16x16x32_bf16(false, a0, false, b1, (short)0, c01, false, false);
    c10 = __builtin_amdgcn_wmma_f32_16x16x32_bf16(false, a1, false, b0, (short)0, c10, false, false);
    c11 = __builtin_amdgcn_wmma_f32_16x16x32_bf16(false, a1, false, b1, (short)0, c11, false, false);
  }
  // D layout: VGPR r holds row m = r + 8*(lane>>4), col n = lane&15
  const int mb = m0 + wm * 32 + (lane >> 4) * 8;
  const int nb = n0 + wn * 32 + (lane & 15);
  auto store_tile = [&](const v8f& c, int mt, int nt) {
    int n = nb + nt * 16;
    #pragma unroll
    for (int r = 0; r < 8; ++r) {
      float v = c[r];
      if (bias) v += bias[n];
      if (ACT == 1) v = fmaxf(v, 0.f);
      size_t o = (size_t)(mb + mt * 16 + r) * N + n;
      if (WBF) C[o] = (__bf16)v;
      if (WF32) cf[o] = v;
    }
  };
  store_tile(c00, 0, 0); store_tile(c01, 0, 1);
  store_tile(c10, 1, 0); store_tile(c11, 1, 1);
}

// split-K reduction: out[m,n] = relu(sum_s part[s,m,n] + bias[n]) as bf16
__global__ void splitk_reduce_relu_kernel(const float* __restrict__ part,
                                          const float* __restrict__ bias,
                                          __bf16* __restrict__ out,
                                          int MN, int N, int S) {
  int i = blockIdx.x * 256 + threadIdx.x;
  if (i >= MN) return;
  float s = bias[i & (N - 1)];
  for (int j = 0; j < S; ++j) s += part[(size_t)j * MN + i];
  out[i] = (__bf16)fmaxf(s, 0.f);
}

// ---------------------------------------------------------------------------
// Fused GAT attention per (b,h): si/sj -> masked leaky_relu scores -> softmax
// (normalization folded into WMMA epilogue) -> attn@hW -> ELU -> bf16 out.
// LDS: 8KB hW^T + 34KB exp-scores + ~3KB reductions = ~45KB.
// ---------------------------------------------------------------------------
#define LD_E 136

__global__ __launch_bounds__(256)
void gat_attn_kernel(const __bf16* __restrict__ hW, const float* __restrict__ a,
                     const float* __restrict__ adj, __bf16* __restrict__ hOut) {
  __shared__ __align__(64) __bf16 sHWt[GHID * Ft];   // [o][y]
  __shared__ __align__(64) __bf16 sE[Ft * LD_E];     // exp(score - rowmax) [x][y]
  __shared__ float sSi[Ft], sSj[Ft], sRed[256], sRinv[Ft];
  const int t = threadIdx.x, lane = t & 31, wave = t >> 5;
  const int h = blockIdx.x, b = blockIdx.y;
  const size_t rowbase = (size_t)b * Ft * C1 + (size_t)h * GHID;

  // 1) stage hW tile transposed: sHWt[o][y] = hW[(b*128+y)*256 + h*32 + o]
  #pragma unroll
  for (int it = 0; it < 16; ++it) {
    int idx = t + 256 * it;                  // 0..4095
    int y = idx >> 5, o = idx & 31;
    sHWt[o * Ft + y] = hW[rowbase + (size_t)y * C1 + o];
  }
  __syncthreads();

  // 2) si, sj
  if (t < Ft) {
    float s = 0.f;
    #pragma unroll
    for (int o = 0; o < GHID; ++o) s += (float)sHWt[o * Ft + t] * a[h * 64 + o];
    sSi[t] = s;
  } else {
    int f = t - Ft; float s = 0.f;
    #pragma unroll
    for (int o = 0; o < GHID; ++o) s += (float)sHWt[o * Ft + f] * a[h * 64 + GHID + o];
    sSj[f] = s;
  }
  __syncthreads();

  // 3) scores: thread pair (x = t>>1) covers one row, 64 cols each half
  const int x = t >> 1, yb = (t & 1) * 64;
  const float si = sSi[x];
  float lmax = -3.0e38f;
  for (int j = 0; j < 64; ++j) {
    int y = yb + j;
    float v = si + sSj[y];
    v = v > 0.f ? v : 0.2f * v;              // leaky_relu(alpha=0.2)
    float m = adj[x * Ft + y];
    float s = v * (m > 0.f ? m : -9.0e15f);  // reference: e * mask, then softmax
    lmax = fmaxf(lmax, s);
  }
  sRed[t] = lmax;
  __syncthreads();
  const float rmax = fmaxf(sRed[x * 2], sRed[x * 2 + 1]);
  float lsum = 0.f;
  for (int j = 0; j < 64; ++j) {
    int y = yb + j;
    float v = si + sSj[y];
    v = v > 0.f ? v : 0.2f * v;
    float m = adj[x * Ft + y];
    float s = v * (m > 0.f ? m : -9.0e15f);
    float e = __expf(s - rmax);
    sE[x * LD_E + y] = (__bf16)e;
    lsum += e;
  }
  __syncthreads();
  sRed[t] = lsum;
  __syncthreads();
  if ((t & 1) == 0) sRinv[x] = 1.f / (sRed[x * 2] + sRed[x * 2 + 1]);
  __syncthreads();

  // 4) hp = softmax(e) @ hW via WMMA; wave w owns M-tile w, both 16-col N tiles
  v8f c0 = {}, c1 = {};
  const int mrow = wave * 16 + (lane & 15);
  #pragma unroll
  for (int kc = 0; kc < 4; ++kc) {
    int k0 = kc * 32;
    v16bf afr = load_frag_A(&sE[mrow * LD_E + k0], lane);
    v16bf b0  = load_frag_B(&sHWt[(0  + (lane & 15)) * Ft + k0], lane);
    v16bf b1  = load_frag_B(&sHWt[(16 + (lane & 15)) * Ft + k0], lane);
    c0 = __builtin_amdgcn_wmma_f32_16x16x32_bf16(false, afr, false, b0, (short)0, c0, false, false);
    c1 = __builtin_amdgcn_wmma_f32_16x16x32_bf16(false, afr, false, b1, (short)0, c1, false, false);
  }
  // epilogue: normalize by 1/rowsum, ELU, store bf16
  const int om = wave * 16 + (lane >> 4) * 8;
  const int oc = lane & 15;
  #pragma unroll
  for (int r = 0; r < 8; ++r) {
    int m = om + r;
    float inv = sRinv[m];
    float v0 = c0[r] * inv, v1 = c1[r] * inv;
    v0 = v0 > 0.f ? v0 : (__expf(v0) - 1.f);   // ELU
    v1 = v1 > 0.f ? v1 : (__expf(v1) - 1.f);
    hOut[rowbase + (size_t)m * C1 + oc]      = (__bf16)v0;
    hOut[rowbase + (size_t)m * C1 + 16 + oc] = (__bf16)v1;
  }
}

// ---------------------------------------------------------------------------
// Final head: y[b] = z2[b,:] . out_w + out_b   (f32)
// ---------------------------------------------------------------------------
__global__ void out_gemv_kernel(const float* __restrict__ z2, const float* __restrict__ w,
                                const float* __restrict__ bptr, float* __restrict__ y) {
  __shared__ float red[256];
  int b = blockIdx.x, t = threadIdx.x;
  red[t] = z2[(size_t)b * MLP_HID + t] * w[t];
  __syncthreads();
  for (int s = 128; s > 0; s >>= 1) {
    if (t < s) red[t] += red[t + s];
    __syncthreads();
  }
  if (t == 0) y[b] = red[0] + bptr[0];
}

// ---------------------------------------------------------------------------
extern "C" void kernel_launch(void* const* d_in, const int* in_sizes, int n_in,
                              void* d_out, int out_size, void* d_ws, size_t ws_size,
                              hipStream_t stream) {
  (void)in_sizes; (void)n_in; (void)out_size; (void)ws_size;
  const int*   ids    = (const int*)  d_in[0];
  const float* vals   = (const float*)d_in[1];
  const float* adj    = (const float*)d_in[2];
  const float* emb    = (const float*)d_in[3];
  const float* W0     = (const float*)d_in[4];
  const float* a0     = (const float*)d_in[5];
  const float* W1     = (const float*)d_in[6];
  const float* a1     = (const float*)d_in[7];
  const float* mlp_w0 = (const float*)d_in[8];
  const float* mlp_b0 = (const float*)d_in[9];
  const float* mlp_w1 = (const float*)d_in[10];
  const float* mlp_b1 = (const float*)d_in[11];
  const float* out_w  = (const float*)d_in[12];
  const float* out_b  = (const float*)d_in[13];
  float* y = (float*)d_out;

  // workspace carve (256B aligned)
  char* p = (char*)d_ws;
  auto alloc = [&](size_t bytes) { char* r = p; p += (bytes + 255) & ~(size_t)255; return r; };
  const size_t NR = (size_t)Bt * Ft;                  // 65536 rows
  const int SPLITK = 16;
  __bf16* xh   = (__bf16*)alloc(NR * NEMB * 2);       // gathered input, bf16
  __bf16* w0t  = (__bf16*)alloc((size_t)C1 * NEMB * 2);       // [256,64]  N-major
  __bf16* w1t  = (__bf16*)alloc((size_t)C1 * C1 * 2);         // [256,256] N-major
  __bf16* mw0t = (__bf16*)alloc((size_t)MLP_HID * DIN * 2);   // [256,32768] N-major
  __bf16* mw1t = (__bf16*)alloc((size_t)MLP_HID * MLP_HID * 2);
  __bf16* hWb  = (__bf16*)alloc(NR * C1 * 2);         // reused for hW0 then hW1
  __bf16* hbuf = (__bf16*)alloc(NR * C1 * 2);         // reused for h1 then h2
  float*  part = (float*) alloc((size_t)SPLITK * Bt * MLP_HID * 4);  // split-K partials
  __bf16* z1   = (__bf16*)alloc((size_t)Bt * MLP_HID * 2);
  __bf16* z2b  = (__bf16*)alloc((size_t)Bt * MLP_HID * 2);
  float*  z2f  = (float*) alloc((size_t)Bt * MLP_HID * 4);

  // 1) prep: gather + weight repack/transpose to N-major bf16
  gather_scale_kernel<<<(NR * NEMB) / 256, 256, 0, stream>>>(ids, vals, emb, xh);
  cat_w_t_kernel<<<(C1 * NEMB + 255) / 256, 256, 0, stream>>>(W0, w0t, NEMB);
  cat_w_t_kernel<<<(C1 * C1 + 255) / 256, 256, 0, stream>>>(W1, w1t, C1);
  trans_f2bf_kernel<<<dim3(DIN / 64, MLP_HID / 64), 256, 0, stream>>>(mlp_w0, mw0t, DIN, MLP_HID);
  trans_f2bf_kernel<<<dim3(MLP_HID / 64, MLP_HID / 64), 256, 0, stream>>>(mlp_w1, mw1t, MLP_HID, MLP_HID);

  // 2) layer-0 projection: hW0 = xh @ W0cat   [65536,64]x[64,256]
  gemm_bf16_kernel<0, true, false><<<dim3(C1 / TBN, NR / TBM, 1), 256, 0, stream>>>(
      xh, w0t, nullptr, hWb, nullptr, (int)NR, C1, NEMB, NEMB);
  // 3) fused attention + ELU -> h1
  gat_attn_kernel<<<dim3(NHEAD, Bt), 256, 0, stream>>>(hWb, a0, adj, hbuf);
  // 4) layer-1 projection: hW1 = h1 @ W1cat   [65536,256]x[256,256]
  gemm_bf16_kernel<0, true, false><<<dim3(C1 / TBN, NR / TBM, 1), 256, 0, stream>>>(
      hbuf, w1t, nullptr, hWb, nullptr, (int)NR, C1, C1, C1);
  // 5) fused attention + ELU -> h2 (overwrites h1; attn reads only hWb)
  gat_attn_kernel<<<dim3(NHEAD, Bt), 256, 0, stream>>>(hWb, a1, adj, hbuf);
  // 6) MLP0 split-K: partials = h2.reshape(512,32768) @ mlp_w0, then reduce+bias+relu
  gemm_bf16_kernel<0, false, true><<<dim3(MLP_HID / TBN, Bt / TBM, SPLITK), 256, 0, stream>>>(
      hbuf, mw0t, nullptr, nullptr, part, Bt, MLP_HID, DIN, DIN / SPLITK);
  splitk_reduce_relu_kernel<<<(Bt * MLP_HID + 255) / 256, 256, 0, stream>>>(
      part, mlp_b0, z1, Bt * MLP_HID, MLP_HID, SPLITK);
  // 7) MLP1: z2 = relu(z1 @ mlp_w1 + b1), keep f32 copy
  gemm_bf16_kernel<1, true, true><<<dim3(MLP_HID / TBN, Bt / TBM, 1), 256, 0, stream>>>(
      z1, mw1t, mlp_b1, z2b, z2f, Bt, MLP_HID, MLP_HID, MLP_HID);
  // 8) output head
  out_gemv_kernel<<<Bt, 256, 0, stream>>>(z2f, out_w, out_b, y);
}
